// TransformerEncoder_53695681134867
// MI455X (gfx1250) — matrix operand order repaired and checked by reading, not compiled, not run on hardware
//
#include <hip/hip_runtime.h>
#include <cmath>

// ---------------- problem constants (match reference) ----------------
constexpr int kB  = 8;
constexpr int kS  = 1024;
constexpr int kIn = 128;
constexpr int kE  = 512;
constexpr int kH  = 8;
constexpr int kFF = 2048;
constexpr int kL  = 4;
constexpr int kHD = 64;            // kE / kH
constexpr float kEps = 1e-5f;

typedef __attribute__((ext_vector_type(16))) _Float16 v16h;
typedef __attribute__((ext_vector_type(8)))  _Float16 v8h;
typedef __attribute__((ext_vector_type(8)))  float    v8f;

// ---------------- WMMA helpers (CDNA5 16x16x32 f16 -> f32) ----------------
__device__ inline v8f wmma_f32_16x16x32(v16h a, v16h b, v8f c) {
  // 8 args: (neg_a, A, neg_b, B, c_mod, C, reuse_a, reuse_b)
  return __builtin_amdgcn_wmma_f32_16x16x32_f16(false, a, false, b, (short)0, c,
                                                false, false);
}

// A fragment: 16x32 f16, row-major source, lda in elements.
// lane L: row = L&15 ; K-base = (L>>4)*16 ; elems 0..15 -> K..K+15
__device__ inline v16h load_a_frag(const _Float16* __restrict__ p0, int lda, int lane) {
  const _Float16* p = p0 + (size_t)(lane & 15) * lda + ((lane >> 4) << 4);
  v8h lo = *(const v8h*)(p);
  v8h hi = *(const v8h*)(p + 8);
  v16h f;
#pragma unroll
  for (int i = 0; i < 8; ++i) { f[i] = lo[i]; f[i + 8] = hi[i]; }
  return f;
}

// B fragment: 32x16 (KxN) where source is row-major [N,K] (weight layout).
// lane L: col = L&15 ; K-base = (L>>4)*16 ; 16 contiguous K values.
__device__ inline v16h load_b_frag(const _Float16* __restrict__ p0, int ldb, int lane) {
  return *(const v16h*)(p0 + (size_t)(lane & 15) * ldb + ((lane >> 4) << 4));
}

// ---------------- generic TN GEMM: C[M,N] = A[M,K] * B[N,K]^T ----------------
// Block: 256 threads = 8 waves in 4x2 grid -> 128x64 tile; each wave 32x32.
__global__ __launch_bounds__(256)
void gemm_tn_kernel(const _Float16* __restrict__ A, int lda, long long strideA,
                    const _Float16* __restrict__ B, int ldb, long long strideB,
                    float* __restrict__ outF32, _Float16* __restrict__ outF16,
                    int ldc, long long strideC,
                    const float* __restrict__ bias, int K, int relu) {
  const int lane = threadIdx.x & 31;
  const int wave = threadIdx.x >> 5;
  const int wrow = wave >> 1, wcol = wave & 1;
  const long long bz = blockIdx.z;
  const _Float16* Ab = A + bz * strideA;
  const _Float16* Bb = B + bz * strideB;
  const int rowBase = blockIdx.y * 128 + wrow * 32;
  const int colBase = blockIdx.x * 64 + wcol * 32;

  v8f acc00 = {}, acc01 = {}, acc10 = {}, acc11 = {};
  for (int k0 = 0; k0 < K; k0 += 32) {
    v16h a0 = load_a_frag(Ab + (size_t)rowBase * lda + k0, lda, lane);
    v16h a1 = load_a_frag(Ab + (size_t)(rowBase + 16) * lda + k0, lda, lane);
    v16h b0 = load_b_frag(Bb + (size_t)colBase * ldb + k0, ldb, lane);
    v16h b1 = load_b_frag(Bb + (size_t)(colBase + 16) * ldb + k0, ldb, lane);
    acc00 = wmma_f32_16x16x32(a0, b0, acc00);
    acc01 = wmma_f32_16x16x32(a0, b1, acc01);
    acc10 = wmma_f32_16x16x32(a1, b0, acc10);
    acc11 = wmma_f32_16x16x32(a1, b1, acc11);
  }

  const int halfsel = (lane >> 4) << 3;   // +8 rows for upper lanes
  const int cn = lane & 15;
#pragma unroll
  for (int rt = 0; rt < 2; ++rt) {
#pragma unroll
    for (int ct = 0; ct < 2; ++ct) {
      const v8f& acc = rt == 0 ? (ct == 0 ? acc00 : acc01) : (ct == 0 ? acc10 : acc11);
      const int col = colBase + ct * 16 + cn;
      const float bval = bias ? bias[col] : 0.0f;
#pragma unroll
      for (int r = 0; r < 8; ++r) {
        const int row = rowBase + rt * 16 + halfsel + r;
        float v = acc[r] + bval;
        if (relu) v = fmaxf(v, 0.0f);
        const size_t idx = (size_t)(bz * strideC) + (size_t)row * ldc + col;
        if (outF32) outF32[idx] = v;
        if (outF16) outF16[idx] = (_Float16)v;
      }
    }
  }
}

// ---------------- fused attention weights ----------------
// grid: (S/16, B); block 256 = 8 waves, wave w owns key cols [w*128, w*128+128)
__device__ inline void red_rows_max(float pr[8], float out[8], float (*red)[16],
                                    int wave, int lane, int halfsel) {
#pragma unroll
  for (int m = 1; m <= 8; m <<= 1)
#pragma unroll
    for (int r = 0; r < 8; ++r)
      pr[r] = fmaxf(pr[r], __shfl_xor(pr[r], m, 16));
  if ((lane & 15) == 0)
#pragma unroll
    for (int r = 0; r < 8; ++r) red[wave][halfsel + r] = pr[r];
  __syncthreads();
#pragma unroll
  for (int r = 0; r < 8; ++r) {
    float v = red[0][halfsel + r];
#pragma unroll
    for (int w = 1; w < 8; ++w) v = fmaxf(v, red[w][halfsel + r]);
    out[r] = v;
  }
  __syncthreads();
}

__device__ inline void red_rows_sum(float pr[8], float out[8], float (*red)[16],
                                    int wave, int lane, int halfsel) {
#pragma unroll
  for (int m = 1; m <= 8; m <<= 1)
#pragma unroll
    for (int r = 0; r < 8; ++r)
      pr[r] += __shfl_xor(pr[r], m, 16);
  if ((lane & 15) == 0)
#pragma unroll
    for (int r = 0; r < 8; ++r) red[wave][halfsel + r] = pr[r];
  __syncthreads();
#pragma unroll
  for (int r = 0; r < 8; ++r) {
    float v = 0.0f;
#pragma unroll
    for (int w = 0; w < 8; ++w) v += red[w][halfsel + r];
    out[r] = v;
  }
  __syncthreads();
}

__global__ __launch_bounds__(256)
void attn_weights_kernel(const _Float16* __restrict__ qk,   // [B,S,2E]  (q | k)
                         const float* __restrict__ db,      // [B,S,S] dist bias
                         _Float16* __restrict__ wout) {     // [B,S,S] f16
  __shared__ float red[8][16];
  const int lane = threadIdx.x & 31;
  const int wave = threadIdx.x >> 5;
  const int qt = blockIdx.x;     // q tile (16 rows)
  const int b  = blockIdx.y;
  const int row0 = qt * 16;
  const int colW0 = wave * 128;  // this wave's 128 key columns
  const int halfsel = (lane >> 4) << 3;
  const int cn = lane & 15;
  const int ld = 2 * kE;         // 1024

  const _Float16* qbase = qk + (size_t)b * kS * ld;

  float accw[8][8];              // head-averaged probabilities [ct][r]
#pragma unroll
  for (int ct = 0; ct < 8; ++ct)
#pragma unroll
    for (int r = 0; r < 8; ++r) accw[ct][r] = 0.0f;

  for (int h = 0; h < kH; ++h) {
    const _Float16* qh = qbase + (size_t)row0 * ld + h * kHD;
    v16h a0 = load_a_frag(qh, ld, lane);
    v16h a1 = load_a_frag(qh + 32, ld, lane);
    const _Float16* kh = qbase + kE + h * kHD;   // k section of the row

    float sc[8][8];
#pragma unroll
    for (int ct = 0; ct < 8; ++ct) {
      const int n0 = colW0 + ct * 16;
      v16h b0 = load_b_frag(kh + (size_t)n0 * ld, ld, lane);
      v16h b1 = load_b_frag(kh + (size_t)n0 * ld + 32, ld, lane);
      v8f c = {};
      c = wmma_f32_16x16x32(a0, b0, c);
      c = wmma_f32_16x16x32(a1, b1, c);
#pragma unroll
      for (int r = 0; r < 8; ++r) sc[ct][r] = c[r] * 0.125f;  // 1/sqrt(64)
    }

    // row max over all 1024 columns
    float pr[8], rmax[8];
#pragma unroll
    for (int r = 0; r < 8; ++r) {
      float m = sc[0][r];
#pragma unroll
      for (int ct = 1; ct < 8; ++ct) m = fmaxf(m, sc[ct][r]);
      pr[r] = m;
    }
    red_rows_max(pr, rmax, red, wave, lane, halfsel);

    // exp + row sum
    float ps[8];
#pragma unroll
    for (int r = 0; r < 8; ++r) ps[r] = 0.0f;
#pragma unroll
    for (int ct = 0; ct < 8; ++ct)
#pragma unroll
      for (int r = 0; r < 8; ++r) {
        float e = __expf(sc[ct][r] - rmax[r]);
        sc[ct][r] = e;
        ps[r] += e;
      }
    float rsum[8];
    red_rows_sum(ps, rsum, red, wave, lane, halfsel);

#pragma unroll
    for (int ct = 0; ct < 8; ++ct)
#pragma unroll
      for (int r = 0; r < 8; ++r)
        accw[ct][r] += sc[ct][r] / rsum[r];      // 1/H factor cancels later
  }

  // distance bias + renormalize
  const float* dbb = db + ((size_t)b * kS + row0) * kS;
#pragma unroll
  for (int ct = 0; ct < 8; ++ct) {
    const int n = colW0 + ct * 16 + cn;
#pragma unroll
    for (int r = 0; r < 8; ++r) {
      const int m = halfsel + r;
      accw[ct][r] *= dbb[(size_t)m * kS + n];
    }
  }
  float ps[8], rsum[8];
#pragma unroll
  for (int r = 0; r < 8; ++r) {
    float s = 0.0f;
#pragma unroll
    for (int ct = 0; ct < 8; ++ct) s += accw[ct][r];
    ps[r] = s;
  }
  red_rows_sum(ps, rsum, red, wave, lane, halfsel);

  _Float16* wr = wout + ((size_t)b * kS + row0) * kS;
#pragma unroll
  for (int ct = 0; ct < 8; ++ct) {
    const int n = colW0 + ct * 16 + cn;
#pragma unroll
    for (int r = 0; r < 8; ++r) {
      const int m = halfsel + r;
      wr[(size_t)m * kS + n] = (_Float16)(accw[ct][r] / rsum[r]);
    }
  }
}

// ---------------- elementwise / reduction kernels ----------------
__global__ void cvt_f16_kernel(const float* __restrict__ in,
                               _Float16* __restrict__ out, int n) {
  int i = blockIdx.x * 256 + threadIdx.x;
  if (i < n) out[i] = (_Float16)in[i];
}

__global__ __launch_bounds__(256)
void bn_stats_kernel(const float* __restrict__ x, float* __restrict__ mean,
                     float* __restrict__ var) {
  __shared__ float rs[256], rs2[256];
  const int c = blockIdx.x;      // channel 0..E-1
  const int t = threadIdx.x;
  float s = 0.0f, s2 = 0.0f;
  for (int r = t; r < kB * kS; r += 256) {
    float v = x[(size_t)r * kE + c];
    s += v; s2 += v * v;
  }
  rs[t] = s; rs2[t] = s2; __syncthreads();
  for (int o = 128; o > 0; o >>= 1) {
    if (t < o) { rs[t] += rs[t + o]; rs2[t] += rs2[t + o]; }
    __syncthreads();
  }
  if (t == 0) {
    float m = rs[0] / (float)(kB * kS);
    mean[c] = m;
    var[c]  = rs2[0] / (float)(kB * kS) - m * m;
  }
}

__global__ __launch_bounds__(256)
void bn_apply_pe_kernel(const float* __restrict__ xin,
                        const float* __restrict__ mean, const float* __restrict__ var,
                        const float* __restrict__ g, const float* __restrict__ bb,
                        float* __restrict__ xout, _Float16* __restrict__ xh,
                        _Float16* __restrict__ xt) {
  size_t i = (size_t)blockIdx.x * 256 + threadIdx.x;   // over B*S*E
  const int c = (int)(i & (kE - 1));
  const size_t row = i >> 9;                            // E = 512
  const int s = (int)(row & (kS - 1));
  const int b = (int)(row >> 10);
  const int p = c >> 1;
  const float dt = expf(-(float)(2 * p) * (logf(10000.0f) / (float)kE));
  const float ang = (float)s * dt;
  const float pe = (c & 1) ? cosf(ang) : sinf(ang);
  float v = (xin[i] - mean[c]) * rsqrtf(var[c] + kEps) * g[c] + bb[c] + pe;
  xout[i] = v;
  xh[i] = (_Float16)v;
  xt[((size_t)b * kE + c) * kS + s] = (_Float16)v;
}

__global__ __launch_bounds__(256)
void dist_bias_kernel(const float* __restrict__ d, const float* __restrict__ scp,
                      float* __restrict__ db) {
  __shared__ float rb[256];
  const int row = blockIdx.x;    // b*S + s
  const int t = threadIdx.x;
  const float* dr = d + (size_t)row * kS;
  float m = -1e30f;
  for (int j = t; j < kS; j += 256) m = fmaxf(m, dr[j]);
  rb[t] = m; __syncthreads();
  for (int o = 128; o > 0; o >>= 1) {
    if (t < o) rb[t] = fmaxf(rb[t], rb[t + o]);
    __syncthreads();
  }
  const float dmax = rb[0];
  const float sc = *scp;
  for (int j = t; j < kS; j += 256)
    db[(size_t)row * kS + j] = expf(sc * (1.0f - dr[j] / dmax));
}

// x = LayerNorm(xin + addv) * g + b ; also emit f16 and transposed f16 copies
__global__ __launch_bounds__(256)
void add_ln_kernel(const float* __restrict__ xin, const float* __restrict__ addv,
                   const float* __restrict__ g, const float* __restrict__ bta,
                   float* __restrict__ xout, _Float16* __restrict__ xh,
                   _Float16* __restrict__ xt) {
  __shared__ float rb[256];
  const int row = blockIdx.x;    // 0..B*S-1
  const int t = threadIdx.x;
  const size_t base = (size_t)row * kE;
  float v0 = xin[base + t] + addv[base + t];
  float v1 = xin[base + t + 256] + addv[base + t + 256];
  rb[t] = v0 + v1; __syncthreads();
  for (int o = 128; o > 0; o >>= 1) { if (t < o) rb[t] += rb[t + o]; __syncthreads(); }
  const float m = rb[0] / (float)kE;
  __syncthreads();
  const float d0 = v0 - m, d1 = v1 - m;
  rb[t] = d0 * d0 + d1 * d1; __syncthreads();
  for (int o = 128; o > 0; o >>= 1) { if (t < o) rb[t] += rb[t + o]; __syncthreads(); }
  const float rstd = rsqrtf(rb[0] / (float)kE + kEps);
  const int b = row >> 10, s = row & (kS - 1);
  const float y0 = d0 * rstd * g[t] + bta[t];
  const float y1 = d1 * rstd * g[t + 256] + bta[t + 256];
  xout[base + t] = y0;        xout[base + t + 256] = y1;
  xh[base + t] = (_Float16)y0; xh[base + t + 256] = (_Float16)y1;
  xt[((size_t)b * kE + t) * kS + s]       = (_Float16)y0;
  xt[((size_t)b * kE + t + 256) * kS + s] = (_Float16)y1;
}

// ---------------- host launch ----------------
extern "C" void kernel_launch(void* const* d_in, const int* in_sizes, int n_in,
                              void* d_out, int out_size, void* d_ws, size_t ws_size,
                              hipStream_t stream) {
  const float* src        = (const float*)d_in[0];
  const float* distances  = (const float*)d_in[1];
  const float* proj_w     = (const float*)d_in[2];
  const float* proj_b     = (const float*)d_in[3];
  const float* bn_g       = (const float*)d_in[4];
  const float* bn_b       = (const float*)d_in[5];
  const float* in_proj_w  = (const float*)d_in[6];
  const float* in_proj_b  = (const float*)d_in[7];
  const float* dist_scale = (const float*)d_in[8];
  const float* lin1_w     = (const float*)d_in[9];
  const float* lin1_b     = (const float*)d_in[10];
  const float* lin2_w     = (const float*)d_in[11];
  const float* lin2_b     = (const float*)d_in[12];
  const float* n1_g       = (const float*)d_in[13];
  const float* n1_b       = (const float*)d_in[14];
  const float* n2_g       = (const float*)d_in[15];
  const float* n2_b       = (const float*)d_in[16];
  float* out = (float*)d_out;

  // workspace carving (256B aligned)
  char* p = (char*)d_ws;
  auto alloc = [&](size_t bytes) -> void* {
    void* r = (void*)p;
    p += (bytes + 255) & ~(size_t)255;
    return r;
  };
  const size_t NBS = (size_t)kB * kS;                 // 8192 rows
  float*    xbuf   = (float*)   alloc(NBS * kE * 4);           // x (f32)
  float*    attn   = (float*)   alloc(NBS * kE * 4);           // attn_out (f32)
  float*    ffo    = (float*)   alloc(NBS * kE * 4);           // ffn out (f32)
  float*    db     = (float*)   alloc((size_t)kB * kS * kS * 4); // dist bias
  float*    meanb  = (float*)   alloc(kE * 4);
  float*    varb   = (float*)   alloc(kE * 4);
  _Float16* src_h  = (_Float16*)alloc(NBS * kIn * 2);
  _Float16* x_h    = (_Float16*)alloc(NBS * kE * 2);
  _Float16* xt_h   = (_Float16*)alloc(NBS * kE * 2);           // [B,E,S]
  _Float16* pw_h   = (_Float16*)alloc((size_t)kE * kIn * 2);
  _Float16* ipw_h  = (_Float16*)alloc((size_t)2 * kE * kE * 2); // q,k rows only
  _Float16* l1w_h  = (_Float16*)alloc((size_t)kFF * kE * 2);
  _Float16* l2w_h  = (_Float16*)alloc((size_t)kE * kFF * 2);
  _Float16* qk_h   = (_Float16*)alloc(NBS * 2 * kE * 2);        // [B,S,2E]
  _Float16* w_h    = (_Float16*)alloc((size_t)kB * kS * kS * 2); // attn weights
  _Float16* hh     = (_Float16*)alloc(NBS * kFF * 2);           // ffn hidden
  (void)ws_size; (void)n_in; (void)in_sizes; (void)out_size;

  auto cvt = [&](const float* in, _Float16* o, int n) {
    cvt_f16_kernel<<<(n + 255) / 256, 256, 0, stream>>>(in, o, n);
  };
  auto gemm = [&](const _Float16* A, int lda, long long sA,
                  const _Float16* Bm, int ldb, long long sB,
                  float* oF, _Float16* oH, int ldc, long long sC,
                  const float* bias, int M, int N, int K, int relu, int batches) {
    dim3 g(N / 64, M / 128, batches);
    gemm_tn_kernel<<<g, 256, 0, stream>>>(A, lda, sA, Bm, ldb, sB, oF, oH,
                                          ldc, sC, bias, K, relu);
  };

  // precompute f16 operands
  cvt(src,       src_h, kB * kS * kIn);
  cvt(proj_w,    pw_h,  kE * kIn);
  cvt(in_proj_w, ipw_h, 2 * kE * kE);       // only q,k rows (v is unused)
  cvt(lin1_w,    l1w_h, kFF * kE);
  cvt(lin2_w,    l2w_h, kE * kFF);

  // input projection: xbuf = src @ proj_w^T + proj_b
  gemm(src_h, kIn, 0, pw_h, kIn, 0, xbuf, nullptr, kE, 0,
       proj_b, kB * kS, kE, kIn, 0, 1);

  // batchnorm (training-mode batch stats) + positional encoding
  bn_stats_kernel<<<kE, 256, 0, stream>>>(xbuf, meanb, varb);
  bn_apply_pe_kernel<<<(int)(NBS * kE / 256), 256, 0, stream>>>(
      xbuf, meanb, varb, bn_g, bn_b, xbuf, x_h, xt_h);

  // distance bias (shared by all layers)
  dist_bias_kernel<<<kB * kS, 256, 0, stream>>>(distances, dist_scale, db);

  for (int l = 0; l < kL; ++l) {
    // q,k projection -> qk_h [B,S,2E] (f16)
    gemm(x_h, kE, 0, ipw_h, kE, 0, nullptr, qk_h, 2 * kE, 0,
         in_proj_b, kB * kS, 2 * kE, kE, 0, 1);

    // fused per-head softmax + head average + dist bias + renorm -> w_h
    attn_weights_kernel<<<dim3(kS / 16, kB), 256, 0, stream>>>(qk_h, db, w_h);

    // attn_out = w @ x   (batched; B operand is x^T so it is K-contiguous)
    gemm(w_h, kS, (long long)kS * kS, xt_h, kS, (long long)kE * kS,
         attn, nullptr, kE, (long long)kS * kE,
         nullptr, kS, kE, kS, 0, kB);

    // x = LN(x + attn_out)
    add_ln_kernel<<<(int)NBS, 256, 0, stream>>>(xbuf, attn, n1_g, n1_b,
                                                xbuf, x_h, xt_h);

    // FFN: hh = relu(x @ lin1^T + b1) ; ffo = hh @ lin2^T + b2
    gemm(x_h, kE, 0, l1w_h, kE, 0, nullptr, hh, kFF, 0,
         lin1_b, kB * kS, kFF, kE, 1, 1);
    gemm(hh, kFF, 0, l2w_h, kFF, 0, ffo, nullptr, kE, 0,
         lin2_b, kB * kS, kE, kFF, 0, 1);

    // x = LN(x + ffn)  (final layer writes straight to d_out)
    float* xo = (l == kL - 1) ? out : xbuf;
    add_ln_kernel<<<(int)NBS, 256, 0, stream>>>(xbuf, ffo, n2_g, n2_b,
                                                xo, x_h, xt_h);
  }
}
